// RnnReceiver_51702816309553
// MI455X (gfx1250) — matrix operand order repaired and checked by reading, not compiled, not run on hardware
//
#include <hip/hip_runtime.h>
#include <hip/hip_bf16.h>
#include <math.h>

// ---------------------------------------------------------------------------
// Types / helpers
// ---------------------------------------------------------------------------
typedef __attribute__((ext_vector_type(16))) __bf16 bf16x16;
typedef __attribute__((ext_vector_type(8)))  float  f32x8;
typedef __attribute__((ext_vector_type(4)))  int    v4i;

union FragBF { bf16x16 v; uint4 q[2]; };

#define AS1 __attribute__((address_space(1)))
#define AS3 __attribute__((address_space(3)))

#if __has_builtin(__builtin_amdgcn_global_load_async_to_lds_b128) && \
    __has_builtin(__builtin_amdgcn_s_wait_asynccnt)
#define HAVE_ASYNC_LDS 1
#else
#define HAVE_ASYNC_LDS 0
#endif

#if HAVE_ASYNC_LDS
// builtin prototype: void (v4i AS1* src, v4i AS3* dst, imm int off, imm int cpol)
#define ASYNC_CP16(gsrc, ldst)                                      \
  __builtin_amdgcn_global_load_async_to_lds_b128(                   \
      (AS1 v4i*)(AS1 void*)(const void*)(gsrc),                     \
      (AS3 v4i*)(AS3 void*)(void*)(ldst), 0, 0)
#endif

__device__ __forceinline__ unsigned short f2bf(float x) {
  union { float f; unsigned u; } v; v.f = x;
  unsigned r = v.u + 0x7FFFu + ((v.u >> 16) & 1u);   // round-to-nearest-even
  return (unsigned short)(r >> 16);
}

// ---------------------------------------------------------------------------
// Tiled WMMA GEMM:  Out[M,N] = A[M,K] * Bmat[N,K]^T  (+ epilogue)
//   MODE 0: A is f32 (converted to bf16 on stage), out f32, += bias
//   MODE 1: A is bf16, out bf16, v = tanh(acc + X[row*xstride+col] + bias)
//   MODE 2: A is bf16, out f32, += bias
// Workgroup: 256 threads = 8 waves; tile 128(M) x 128(N); K-step 32.
// Double-buffered LDS, 1 barrier / k-step. Tile staging uses
// GLOBAL_LOAD_ASYNC_TO_LDS (zero staging VGPRs, ASYNCcnt-tracked) for bf16
// sources; MODE 0's A tile stages through registers (f32->bf16 convert).
// B fragments are loaded just-in-time (keeps live VGPRs ~100, no spills).
// ---------------------------------------------------------------------------
#define TILE_M 128
#define TILE_N 128
#define TILE_K 32
#define LDS_STRIDE 40   // bf16 elems/row: 32 data + 8 pad (80B, keeps 16B align)

template <int MODE>
__global__ __launch_bounds__(256) void gemm_wmma(
    const void* __restrict__ Asrc,
    const unsigned short* __restrict__ Bmat,   // [N x K] bf16 bits (row-major)
    const float* __restrict__ bias,            // [N]
    const float* __restrict__ X,               // addend (MODE 1), row stride xstride
    long xstride,
    void* __restrict__ Out,
    int M, int N, int K)
{
  __shared__ __align__(16) unsigned short ldsA[2][TILE_M * LDS_STRIDE];
  __shared__ __align__(16) unsigned short ldsB[2][TILE_N * LDS_STRIDE];

  const int tid   = threadIdx.x;
  const int lane  = tid & 31;          // wave32
  const int wave  = tid >> 5;          // 0..7
  const int mtile = blockIdx.y * TILE_M;
  const int ntile = blockIdx.x * TILE_N;

  const int half = (lane < 16) ? 0 : 1;
  const int lmod = lane & 15;

  f32x8 acc[8];
  #pragma unroll
  for (int i = 0; i < 8; ++i) {
    #pragma unroll
    for (int j = 0; j < 8; ++j) acc[i][j] = 0.0f;
  }

  // --- MODE 0 A-tile staging registers (f32 -> bf16 conversion path) ---
  float4 afr[4];
  // --- fallback (no async builtin) staging registers ---
  uint4 aur[2];
  uint4 bur[2];

  // Stage B tile (always bf16): async global->LDS if available.
  auto stageB_issue = [&](int k0, int buf) {
    #pragma unroll
    for (int j = 0; j < 2; ++j) {
      int idx = tid + 256 * j;               // 512 16B chunks
      int row = idx >> 2;
      int c   = idx & 3;
      const unsigned short* g = Bmat + (size_t)(ntile + row) * K + k0 + c * 8;
      unsigned short* l = &ldsB[buf][row * LDS_STRIDE + c * 8];
#if HAVE_ASYNC_LDS
      ASYNC_CP16(g, l);
#else
      bur[j] = *(const uint4*)g;
      (void)l;
#endif
    }
  };
  auto stageB_commit = [&](int buf) {
#if !HAVE_ASYNC_LDS
    #pragma unroll
    for (int j = 0; j < 2; ++j) {
      int idx = tid + 256 * j;
      int row = idx >> 2;
      int c   = idx & 3;
      *(uint4*)&ldsB[buf][row * LDS_STRIDE + c * 8] = bur[j];
    }
#else
    (void)buf;
#endif
  };

  // Stage A tile.
  auto stageA_issue = [&](int k0, int buf) {
    if (MODE == 0) {
      const float* A = (const float*)Asrc;
      #pragma unroll
      for (int j = 0; j < 4; ++j) {
        int idx = tid + 256 * j;             // 1024 float4 chunks
        int row = idx >> 3;
        int c4  = idx & 7;
        afr[j] = *(const float4*)(A + (size_t)(mtile + row) * K + k0 + c4 * 4);
      }
      (void)buf;
    } else {
      const unsigned short* A = (const unsigned short*)Asrc;
      #pragma unroll
      for (int j = 0; j < 2; ++j) {
        int idx = tid + 256 * j;             // 512 16B chunks
        int row = idx >> 2;
        int c   = idx & 3;
        const unsigned short* g = A + (size_t)(mtile + row) * K + k0 + c * 8;
        unsigned short* l = &ldsA[buf][row * LDS_STRIDE + c * 8];
#if HAVE_ASYNC_LDS
        ASYNC_CP16(g, l);
#else
        aur[j] = *(const uint4*)g;
        (void)l;
#endif
      }
    }
  };
  auto stageA_commit = [&](int buf) {
    if (MODE == 0) {
      #pragma unroll
      for (int j = 0; j < 4; ++j) {
        int idx = tid + 256 * j;
        int row = idx >> 3;
        int c4  = idx & 7;
        union { unsigned short h[4]; uint2 u; } pk;
        pk.h[0] = f2bf(afr[j].x); pk.h[1] = f2bf(afr[j].y);
        pk.h[2] = f2bf(afr[j].z); pk.h[3] = f2bf(afr[j].w);
        *(uint2*)&ldsA[buf][row * LDS_STRIDE + c4 * 4] = pk.u;
      }
    } else {
#if !HAVE_ASYNC_LDS
      #pragma unroll
      for (int j = 0; j < 2; ++j) {
        int idx = tid + 256 * j;
        int row = idx >> 2;
        int c   = idx & 3;
        *(uint4*)&ldsA[buf][row * LDS_STRIDE + c * 8] = aur[j];
      }
#else
      (void)buf;
#endif
    }
  };

  // --- prologue: stage tile 0 into buffer 0 ---
  stageA_issue(0, 0);
  stageB_issue(0, 0);
  stageA_commit(0);
  stageB_commit(0);
#if HAVE_ASYNC_LDS
  __builtin_amdgcn_s_wait_asynccnt(0);
#endif
  __syncthreads();

  const int NK = K / TILE_K;
  int p = 0;
  for (int ki = 0; ki < NK; ++ki) {
    const bool hasNext = (ki + 1 < NK);
    // kick off next tile (asyncs / global loads overlap the WMMA burst)
    if (hasNext) {
      stageA_issue((ki + 1) * TILE_K, p ^ 1);
      stageB_issue((ki + 1) * TILE_K, p ^ 1);
    }

    // ---- fragments from lds[p] ----
    const unsigned short* LA = ldsA[p];
    const unsigned short* LB = ldsB[p];

    FragBF fa;   // lane L<16 -> K{0..7,16..23}; L>=16 -> K{8..15,24..31}
    {
      const unsigned short* abase = &LA[(wave * 16 + lmod) * LDS_STRIDE];
      fa.q[0] = *(const uint4*)(abase + half * 8);
      fa.q[1] = *(const uint4*)(abase + 16 + half * 8);
    }

    // 8 WMMAs; B fragments just-in-time (low register pressure, no spills).
    // reuse_a/reuse_b must be literal constants, so unroll via macro.
#define WMMA_STEP(n, RA)                                                   \
    {                                                                      \
      FragBF fb;                                                           \
      const unsigned short* bb =                                           \
          &LB[((n) * 16 + lmod) * LDS_STRIDE + half * 8];                  \
      fb.q[0] = *(const uint4*)(bb);                                       \
      fb.q[1] = *(const uint4*)(bb + 16);                                  \
      acc[n] = __builtin_amdgcn_wmma_f32_16x16x32_bf16(                    \
          false, fa.v, false, fb.v, (short)0, acc[n], RA, false);          \
    }
    WMMA_STEP(0, true);
    WMMA_STEP(1, true);
    WMMA_STEP(2, true);
    WMMA_STEP(3, true);
    WMMA_STEP(4, true);
    WMMA_STEP(5, true);
    WMMA_STEP(6, true);
    WMMA_STEP(7, false);
#undef WMMA_STEP

    // ---- finalize next tile; single barrier per iteration ----
    if (hasNext) {
      stageA_commit(p ^ 1);
      stageB_commit(p ^ 1);
#if HAVE_ASYNC_LDS
      __builtin_amdgcn_s_wait_asynccnt(0);
#endif
      __syncthreads();
      p ^= 1;
    }
  }

  // ---- epilogue: C/D layout: VGPR j -> row j + 8*half, lane%16 -> col ----
  const int rowHalf = half * 8;
  #pragma unroll
  for (int n = 0; n < 8; ++n) {
    int col = ntile + n * 16 + lmod;
    float bv = bias[col];
    #pragma unroll
    for (int j = 0; j < 8; ++j) {
      int row = mtile + wave * 16 + rowHalf + j;
      float v = acc[n][j] + bv;
      if (MODE == 1) {
        v += X[(size_t)row * (size_t)xstride + col];
        v = tanhf(v);
        ((unsigned short*)Out)[(size_t)row * N + col] = f2bf(v);
      } else {
        ((float*)Out)[(size_t)row * N + col] = v;
      }
    }
  }
}

// ---------------------------------------------------------------------------
// Prep kernels
// ---------------------------------------------------------------------------
// Wc[h][v] = sum_e W_ih[h][e] * W_emb[e][v]   (1024x1024 bf16)
__global__ __launch_bounds__(256) void make_wc(
    const float* __restrict__ W_ih,    // [1024 x 512]
    const float* __restrict__ W_emb,   // [512 x 1024]
    unsigned short* __restrict__ Wc)   // [1024 x 1024] bf16
{
  __shared__ float wrow[512];
  int h = blockIdx.x;
  for (int e = threadIdx.x; e < 512; e += 256) wrow[e] = W_ih[h * 512 + e];
  __syncthreads();
  for (int v = threadIdx.x; v < 1024; v += 256) {
    float acc = 0.0f;
    for (int e = 0; e < 512; ++e) acc += wrow[e] * W_emb[e * 1024 + v];
    Wc[h * 1024 + v] = f2bf(acc);
  }
}

// bc[h] = b_ih[h] + sum_e W_ih[h][e] * b_emb[e]
__global__ void make_bc(const float* __restrict__ W_ih,
                        const float* __restrict__ b_emb,
                        const float* __restrict__ b_ih,
                        float* __restrict__ bc)
{
  int h = blockIdx.x * blockDim.x + threadIdx.x;
  if (h < 1024) {
    float acc = b_ih[h];
    for (int e = 0; e < 512; ++e) acc += W_ih[h * 512 + e] * b_emb[e];
    bc[h] = acc;
  }
}

__global__ void cvt_f32_bf16(const float* __restrict__ in,
                             unsigned short* __restrict__ out, int n)
{
  int i4 = (blockIdx.x * blockDim.x + threadIdx.x) * 4;
  if (i4 < n) {
    float4 f = *(const float4*)(in + i4);
    union { unsigned short h[4]; uint2 u; } p;
    p.h[0] = f2bf(f.x); p.h[1] = f2bf(f.y);
    p.h[2] = f2bf(f.z); p.h[3] = f2bf(f.w);
    *(uint2*)(out + i4) = p.u;
  }
}

__global__ void zero_u16(unsigned short* __restrict__ p, int n) {
  int i8 = (blockIdx.x * blockDim.x + threadIdx.x) * 8;
  if (i8 < n) {
    uint4 z; z.x = 0; z.y = 0; z.z = 0; z.w = 0;
    *(uint4*)(p + i8) = z;
  }
}

// ---------------------------------------------------------------------------
// Launch
// ---------------------------------------------------------------------------
extern "C" void kernel_launch(void* const* d_in, const int* in_sizes, int n_in,
                              void* d_out, int out_size, void* d_ws, size_t ws_size,
                              hipStream_t stream) {
  (void)in_sizes; (void)n_in; (void)out_size; (void)ws_size;

  const float* msg  = (const float*)d_in[0];   // [1024,32,1024]
  const float* Wemb = (const float*)d_in[1];   // [512,1024]
  const float* bemb = (const float*)d_in[2];   // [512]
  const float* Wih  = (const float*)d_in[3];   // [1024,512]
  const float* bih  = (const float*)d_in[4];   // [1024]
  const float* Whh  = (const float*)d_in[5];   // [1024,1024]
  const float* bhh  = (const float*)d_in[6];   // [1024]
  const float* Wout = (const float*)d_in[7];   // [1024,1024]
  const float* bout = (const float*)d_in[8];   // [1024]

  // Workspace carve (256B aligned sections)
  char* ws = (char*)d_ws;
  size_t off = 0;
  auto carve = [&](size_t bytes) -> void* {
    void* p = ws + off;
    off += (bytes + 255) & ~(size_t)255;
    return p;
  };
  float*          xproj = (float*)         carve((size_t)32768 * 1024 * 4); // 134 MB
  unsigned short* wc    = (unsigned short*)carve((size_t)1024 * 1024 * 2);
  unsigned short* whh   = (unsigned short*)carve((size_t)1024 * 1024 * 2);
  unsigned short* wout  = (unsigned short*)carve((size_t)1024 * 1024 * 2);
  unsigned short* h0    = (unsigned short*)carve((size_t)1024 * 1024 * 2);
  unsigned short* h1    = (unsigned short*)carve((size_t)1024 * 1024 * 2);
  float*          bc    = (float*)         carve((size_t)1024 * 4);

  // --- prep: fused weight Wc = W_ih @ W_emb, combined bias, bf16 weights ---
  make_wc<<<1024, 256, 0, stream>>>(Wih, Wemb, wc);
  make_bc<<<4, 256, 0, stream>>>(Wih, bemb, bih, bc);
  cvt_f32_bf16<<<(1024 * 1024 / 4 + 255) / 256, 256, 0, stream>>>(Whh, whh, 1024 * 1024);
  cvt_f32_bf16<<<(1024 * 1024 / 4 + 255) / 256, 256, 0, stream>>>(Wout, wout, 1024 * 1024);
  zero_u16<<<(1024 * 1024 / 8 + 255) / 256, 256, 0, stream>>>(h0, 1024 * 1024);

  // --- x_proj[b*T+t][h] = msg @ Wc^T + bc   (32768 x 1024 x 1024) ---
  gemm_wmma<0><<<dim3(1024 / TILE_N, 32768 / TILE_M), 256, 0, stream>>>(
      msg, wc, bc, nullptr, 0, xproj, 32768, 1024, 1024);

  // --- RNN scan: h = tanh(x_t + h @ Whh^T + bhh), 32 steps ---
  const unsigned short* hin = h0;
  unsigned short* hout = h1;
  for (int t = 0; t < 32; ++t) {
    gemm_wmma<1><<<dim3(1024 / TILE_N, 1024 / TILE_M), 256, 0, stream>>>(
        hin, whh, bhh, xproj + (size_t)t * 1024, (long)32 * 1024,
        hout, 1024, 1024, 1024);
    const unsigned short* tmp = hout;
    hout = (unsigned short*)hin;
    hin = tmp;
  }

  // --- out = h_T @ Wout^T + bout ---
  gemm_wmma<2><<<dim3(1024 / TILE_N, 1024 / TILE_M), 256, 0, stream>>>(
      hin, wout, bout, nullptr, 0, (float*)d_out, 1024, 1024, 1024);
}
